// LSTM_42314017800900
// MI455X (gfx1250) — matrix operand order repaired and checked by reading, not compiled, not run on hardware
//
#include <hip/hip_runtime.h>

#define F_DIM   128
#define IN_DIM  256
#define H_DIM   512
#define BATCH   128
#define SEQ_L   1024
#define NWG0    16            // workgroups owning layer 0 (+ encoder)
#define NWG     32            // total workgroups (16 also own layer 1 + decoder)
#define HSZ     (BATCH * H_DIM)   // elements in one h buffer

typedef __bf16 bf16_t;
typedef __attribute__((ext_vector_type(16))) __bf16 v16bf;
typedef __attribute__((ext_vector_type(8)))  __bf16 v8bf;
typedef __attribute__((ext_vector_type(8)))  float  v8f;

union FragU { v16bf v; v8bf h[2]; };

// ---- WMMA fragment loaders (CDNA5 wave32 layouts, 16x16x32 bf16) ----------
// A (16xK, lane = M row): lane<16 holds K {0..7,16..23}, lane>=16 {8..15,24..31}
__device__ __forceinline__ v16bf load_a_bf16(const bf16_t* base, int stride,
                                             int row0, int k0, int lane) {
  const bf16_t* p = base + (size_t)(row0 + (lane & 15)) * stride
                         + (k0 + ((lane >> 4) << 3));
  FragU u;
  u.h[0] = *(const v8bf*)(p);
  u.h[1] = *(const v8bf*)(p + 16);
  return u.v;
}
// B (Kx16, lane = N col): lanes 0-15 hold K {0..7,16..23}, lanes 16-31 {8..15,24..31}.
// B[k,n] = W[n,k] (row-major W) -> two contiguous 16B runs per lane, no transpose.
__device__ __forceinline__ v16bf load_b_bf16(const bf16_t* base, int stride,
                                             int row0, int k0, int lane) {
  const bf16_t* p = base + (size_t)(row0 + (lane & 15)) * stride
                         + (k0 + ((lane >> 4) << 3));
  FragU u;
  u.h[0] = *(const v8bf*)(p);
  u.h[1] = *(const v8bf*)(p + 16);
  return u.v;
}
// A fragment from fp32 source (encoder input x), converted on the fly
__device__ __forceinline__ v16bf load_a_f32(const float* rowp, int k0, int lane) {
  int k = k0 + ((lane >> 4) << 3);
  FragU u;
#pragma unroll
  for (int j = 0; j < 8; ++j) u.h[0][j] = (__bf16)rowp[k + j];
#pragma unroll
  for (int j = 0; j < 8; ++j) u.h[1][j] = (__bf16)rowp[k + 16 + j];
  return u.v;
}

__device__ __forceinline__ v8f wmma_bf16(v16bf a, v16bf b, v8f c) {
  return __builtin_amdgcn_wmma_f32_16x16x32_bf16(false, a, false, b,
                                                 (short)0, c, false, false);
}

// C/D tile store: lane%16 = N, M = (lane/16)*8 + r  -> coalesced 32B runs
__device__ __forceinline__ void store_tile_bf16(bf16_t* base, int stride,
                                                int row0, int col0, int lane, v8f v) {
  int n  = col0 + (lane & 15);
  int r0 = row0 + ((lane >> 4) << 3);
#pragma unroll
  for (int r = 0; r < 8; ++r) base[(size_t)(r0 + r) * stride + n] = (bf16_t)v[r];
}

__device__ __forceinline__ float sigm_f(float x) { return 1.0f / (1.0f + __expf(-x)); }
__device__ __forceinline__ float tanh_f(float x) { return 1.0f - 2.0f / (1.0f + __expf(2.0f * x)); }

// ---- grid-wide sense barrier (device-scope atomics in L2) ------------------
__device__ __forceinline__ void grid_barrier(int* cnt, int* gen, int phase) {
  __threadfence();          // flush this thread's stores to device scope
  __syncthreads();
  if (threadIdx.x == 0) {
    int old = __hip_atomic_fetch_add(cnt, 1, __ATOMIC_ACQ_REL, __HIP_MEMORY_SCOPE_AGENT);
    if (old == NWG - 1) {
      __hip_atomic_store(cnt, 0, __ATOMIC_RELAXED, __HIP_MEMORY_SCOPE_AGENT);
      __hip_atomic_store(gen, phase + 1, __ATOMIC_RELEASE, __HIP_MEMORY_SCOPE_AGENT);
    } else {
      while (__hip_atomic_load(gen, __ATOMIC_ACQUIRE, __HIP_MEMORY_SCOPE_AGENT) <= phase)
        __builtin_amdgcn_s_sleep(2);
    }
  }
  __syncthreads();
  __threadfence();          // acquire: invalidate near caches before reloading h/e
}

// ---- gate GEMM: acc[g][s] += A[16 x 32k] * W[g*512+wg*32+s*16 .. +16, k]^T --
__device__ __forceinline__ void gemm_gates(v8f (&acc)[4][2],
                                           const bf16_t* A, int astride, int m0,
                                           const bf16_t* W, int wstride,
                                           int wg, int nk, int lane) {
  for (int kc = 0; kc < nk; ++kc) {
    v16bf a = load_a_bf16(A, astride, m0, kc * 32, lane);
#pragma unroll
    for (int g = 0; g < 4; ++g)
#pragma unroll
      for (int s = 0; s < 2; ++s) {
        v16bf b = load_b_bf16(W, wstride, g * H_DIM + wg * 32 + s * 16, kc * 32, lane);
        acc[g][s] = wmma_bf16(a, b, acc[g][s]);   // reuse A across 8 WMMAs
      }
  }
}

// ---- LSTM pointwise update; c stays in VGPRs for the whole sequence --------
__device__ __forceinline__ void lstm_update(v8f (&acc)[4][2], v8f (&c)[2],
                                            const float (&bias)[4][2],
                                            bf16_t* hout, int m0, int wg, int lane) {
#pragma unroll
  for (int s = 0; s < 2; ++s) {
    v8f hv;
#pragma unroll
    for (int r = 0; r < 8; ++r) {
      float iv = acc[0][s][r] + bias[0][s];
      float fv = acc[1][s][r] + bias[1][s];
      float gv = acc[2][s][r] + bias[2][s];
      float ov = acc[3][s][r] + bias[3][s];
      float cn = sigm_f(fv) * c[s][r] + sigm_f(iv) * tanh_f(gv);
      c[s][r]  = cn;
      hv[r]    = sigm_f(ov) * tanh_f(cn);
    }
    store_tile_bf16(hout, H_DIM, m0, wg * 32 + s * 16, lane, hv);
  }
}

// ---- weight fp32 -> bf16 ----------------------------------------------------
__global__ void cvt_bf16_kernel(const float* __restrict__ src, bf16_t* __restrict__ dst, int n) {
  int i = blockIdx.x * blockDim.x + threadIdx.x;
  if (i < n) dst[i] = (bf16_t)src[i];
}

// ---- init: zero h buffers, reset barrier, copy out[:,0,:] = x[:,0,:] -------
__global__ void init_kernel(const float* __restrict__ x, float* __restrict__ out,
                            bf16_t* __restrict__ hzero, int* __restrict__ bar) {
  int i = blockIdx.x * blockDim.x + threadIdx.x;
  if (i < 4 * HSZ) hzero[i] = (bf16_t)0.0f;                 // h0[2] + h1[2]
  if (i < BATCH * F_DIM) {
    int b = i >> 7, f = i & 127;
    out[(size_t)b * SEQ_L * F_DIM + f] = x[(size_t)b * SEQ_L * F_DIM + f];
  }
  if (i < 2) bar[i] = 0;
}

// ---- persistent fused encoder + 2xLSTM + decoder ---------------------------
__global__ void __launch_bounds__(256)
lstm_persistent(const float* __restrict__ x,
                const float* __restrict__ b_e,
                const float* __restrict__ b_ih0, const float* __restrict__ b_hh0,
                const float* __restrict__ b_ih1, const float* __restrict__ b_hh1,
                const float* __restrict__ b_d,
                const bf16_t* __restrict__ wWe,  const bf16_t* __restrict__ wWih0,
                const bf16_t* __restrict__ wWhh0, const bf16_t* __restrict__ wWih1,
                const bf16_t* __restrict__ wWhh1, const bf16_t* __restrict__ wWd,
                bf16_t* __restrict__ e_buf, bf16_t* __restrict__ h0_buf,
                bf16_t* __restrict__ h1_buf, float* __restrict__ out, int* bar) {
  const int lane = threadIdx.x & 31;
  const int w    = threadIdx.x >> 5;      // wave 0..7 -> batch tile
  const bool grp0 = blockIdx.x < NWG0;
  const int wg   = grp0 ? blockIdx.x : blockIdx.x - NWG0;  // hidden slice 32*wg
  const int m0   = w * 16;
  int* bar_cnt = bar;
  int* bar_gen = bar + 1;

  // hoist biases (wave-invariant across time)
  float bias[4][2];
  {
    const float* bi = grp0 ? b_ih0 : b_ih1;
    const float* bh = grp0 ? b_hh0 : b_hh1;
#pragma unroll
    for (int g = 0; g < 4; ++g)
#pragma unroll
      for (int s = 0; s < 2; ++s) {
        int n = g * H_DIM + wg * 32 + s * 16 + (lane & 15);
        bias[g][s] = bi[n] + bh[n];
      }
  }
  float be = grp0 ? b_e[wg * 16 + (lane & 15)] : 0.0f;
  float bd = (!grp0 && wg < 8) ? b_d[wg * 16 + (lane & 15)] : 0.0f;

  v8f c[2] = {};          // persistent cell state tile (c0 for grp0, c1 for grp1)
  int phase = 0;

  for (int t = 0; t < SEQ_L; ++t) {
    // ================= Phase 1 =================
    if (grp0) {
      if (t <= SEQ_L - 2) {       // encoder e(t) = x[:,t,:] @ W_e^T + b_e
        v8f acc = {};
        const float* xrow = x + ((size_t)(m0 + (lane & 15)) * SEQ_L + t) * F_DIM;
        __builtin_prefetch(xrow + F_DIM, 0, 1);   // next timestep of x
#pragma unroll
        for (int kc = 0; kc < F_DIM / 32; ++kc) {
          v16bf a = load_a_f32(xrow, kc * 32, lane);
          v16bf b = load_b_bf16(wWe, F_DIM, wg * 16, kc * 32, lane);
          acc = wmma_bf16(a, b, acc);
        }
#pragma unroll
        for (int r = 0; r < 8; ++r) acc[r] += be;
        store_tile_bf16(e_buf, IN_DIM, m0, wg * 16, lane, acc);
      }
    } else if (t >= 1) {          // layer 1 for step t-1
      const bf16_t* h0p = h0_buf + (size_t)((t + 1) & 1) * HSZ;   // h0(t-1)
      const bf16_t* h1p = h1_buf + (size_t)(t & 1) * HSZ;         // h1(t-2)
      bf16_t*       h1c = h1_buf + (size_t)((t + 1) & 1) * HSZ;   // h1(t-1)
      v8f acc[4][2] = {};
      gemm_gates(acc, h0p, H_DIM, m0, wWih1, H_DIM, wg, H_DIM / 32, lane);
      gemm_gates(acc, h1p, H_DIM, m0, wWhh1, H_DIM, wg, H_DIM / 32, lane);
      lstm_update(acc, c, bias, h1c, m0, wg, lane);
    }
    grid_barrier(bar_cnt, bar_gen, phase++);

    // ================= Phase 2 =================
    if (grp0) {
      if (t <= SEQ_L - 2) {       // layer 0 for step t
        const bf16_t* h0p = h0_buf + (size_t)((t + 1) & 1) * HSZ; // h0(t-1)
        bf16_t*       h0c = h0_buf + (size_t)(t & 1) * HSZ;       // h0(t)
        v8f acc[4][2] = {};
        gemm_gates(acc, e_buf, IN_DIM, m0, wWih0, IN_DIM, wg, IN_DIM / 32, lane);
        gemm_gates(acc, h0p,   H_DIM,  m0, wWhh0, H_DIM,  wg, H_DIM / 32, lane);
        lstm_update(acc, c, bias, h0c, m0, wg, lane);
      }
    } else if (t >= 1 && wg < 8) { // decoder y(t-1) -> out[:, t, :]
      const bf16_t* h1p = h1_buf + (size_t)((t + 1) & 1) * HSZ;   // h1(t-1)
      v8f acc = {};
      for (int kc = 0; kc < H_DIM / 32; ++kc) {
        v16bf a = load_a_bf16(h1p, H_DIM, m0, kc * 32, lane);
        v16bf b = load_b_bf16(wWd, H_DIM, wg * 16, kc * 32, lane);
        acc = wmma_bf16(a, b, acc);
      }
      int n  = wg * 16 + (lane & 15);
      int r0 = m0 + ((lane >> 4) << 3);
#pragma unroll
      for (int r = 0; r < 8; ++r)
        out[((size_t)(r0 + r) * SEQ_L + t) * F_DIM + n] = acc[r] + bd;
    }
    grid_barrier(bar_cnt, bar_gen, phase++);
  }
}

extern "C" void kernel_launch(void* const* d_in, const int* in_sizes, int n_in,
                              void* d_out, int out_size, void* d_ws, size_t ws_size,
                              hipStream_t stream) {
  const float* x     = (const float*)d_in[0];
  const float* W_e   = (const float*)d_in[1];
  const float* b_e   = (const float*)d_in[2];
  const float* W_ih0 = (const float*)d_in[3];
  const float* W_hh0 = (const float*)d_in[4];
  const float* b_ih0 = (const float*)d_in[5];
  const float* b_hh0 = (const float*)d_in[6];
  const float* W_ih1 = (const float*)d_in[7];
  const float* W_hh1 = (const float*)d_in[8];
  const float* b_ih1 = (const float*)d_in[9];
  const float* b_hh1 = (const float*)d_in[10];
  const float* W_d   = (const float*)d_in[11];
  const float* b_d   = (const float*)d_in[12];
  float* out = (float*)d_out;

  // workspace layout
  char* ws = (char*)d_ws;
  int*    bar   = (int*)ws;                      // 2 ints (256B slot)
  bf16_t* h0    = (bf16_t*)(ws + 256);           // 2 * HSZ
  bf16_t* h1    = h0 + 2 * HSZ;                  // 2 * HSZ
  bf16_t* e     = h1 + 2 * HSZ;                  // BATCH * IN_DIM
  bf16_t* wWe   = e    + (size_t)BATCH * IN_DIM;
  bf16_t* wWih0 = wWe  + (size_t)IN_DIM * F_DIM;
  bf16_t* wWhh0 = wWih0 + (size_t)4 * H_DIM * IN_DIM;
  bf16_t* wWih1 = wWhh0 + (size_t)4 * H_DIM * H_DIM;
  bf16_t* wWhh1 = wWih1 + (size_t)4 * H_DIM * H_DIM;
  bf16_t* wWd   = wWhh1 + (size_t)4 * H_DIM * H_DIM;

  auto cvt = [&](const float* s, bf16_t* d, int n) {
    cvt_bf16_kernel<<<(n + 255) / 256, 256, 0, stream>>>(s, d, n);
  };
  cvt(W_e,   wWe,   IN_DIM * F_DIM);
  cvt(W_ih0, wWih0, 4 * H_DIM * IN_DIM);
  cvt(W_hh0, wWhh0, 4 * H_DIM * H_DIM);
  cvt(W_ih1, wWih1, 4 * H_DIM * H_DIM);
  cvt(W_hh1, wWhh1, 4 * H_DIM * H_DIM);
  cvt(W_d,   wWd,   F_DIM * H_DIM);

  init_kernel<<<(4 * HSZ + 255) / 256, 256, 0, stream>>>(x, out, h0, bar);

  lstm_persistent<<<NWG, 256, 0, stream>>>(
      x, b_e, b_ih0, b_hh0, b_ih1, b_hh1, b_d,
      wWe, wWih0, wWhh0, wWih1, wWhh1, wWd,
      e, h0, h1, out, bar);
}
